// ParallelRetention_18107582120671
// MI455X (gfx1250) — compile-verified, matrix operand order checked by reading, not run on hardware
//
#include <hip/hip_runtime.h>
#include <hip/hip_bf16.h>
#include <stdint.h>

// ---------------------------------------------------------------------------
// ParallelRetention on MI455X (gfx1250): bf16 WMMA pipeline, double-buffered
// async-LDS staging.
//   xb  = bf16(x)
//   q/k/v = xb @ W{q,k,v} + b          (WMMA, bf16 out)
//   sb  = bf16(d_gamma * (q @ k^T))    (WMMA, gamma epilogue)
//   rb  = sb @ v                       (WMMA, v pre-transposed)
//   out = PReLU(rb @ Wr + br)          (WMMA, f32 out)
// K-tile i+1 is DMA'd into LDS (GLOBAL_LOAD_ASYNC_TO_LDS_B128, ASYNCcnt)
// while WMMA consumes K-tile i; one s_wait_asynccnt + one barrier per step.
// ---------------------------------------------------------------------------

typedef __bf16 bf16_t;
typedef __attribute__((ext_vector_type(16))) __bf16 v16bf;
typedef __attribute__((ext_vector_type(8)))  __bf16 v8bf;
typedef __attribute__((ext_vector_type(8)))  float  v8f;

#define T_DIM     4096
#define IN_DIM    1024
#define INTER_DIM 1024
#define OUT_DIM   1024

#define USE_ASYNC_LDS 1

// ---------------------------------------------------------------------------
// f32 -> bf16 elementwise convert (4 elements / thread, float4 loads)
// ---------------------------------------------------------------------------
__global__ __launch_bounds__(256)
void convert_f32_to_bf16(const float* __restrict__ in, bf16_t* __restrict__ out, int n4) {
  int i = blockIdx.x * blockDim.x + threadIdx.x;
  if (i >= n4) return;
  float4 f = ((const float4*)in)[i];
  bf16_t* ob = out + (size_t)i * 4;
  ob[0] = (bf16_t)f.x; ob[1] = (bf16_t)f.y; ob[2] = (bf16_t)f.z; ob[3] = (bf16_t)f.w;
}

// ---------------------------------------------------------------------------
// Tiled transpose + convert to bf16:  in[rows x cols] -> out[cols x rows]
// Block (32, 8), grid (cols/32, rows/32). 32x33 LDS tile kills bank conflicts.
// ---------------------------------------------------------------------------
template <typename TIN>
__global__ __launch_bounds__(256)
void transpose_convert_bf16(const TIN* __restrict__ in, bf16_t* __restrict__ out,
                            int rows, int cols) {
  __shared__ float tile[32][33];
  int bx = blockIdx.x * 32;   // col base in `in`
  int by = blockIdx.y * 32;   // row base in `in`
  int tx = threadIdx.x;       // 0..31
  int ty = threadIdx.y;       // 0..7
  for (int i = 0; i < 32; i += 8)
    tile[ty + i][tx] = (float)in[(size_t)(by + ty + i) * cols + (bx + tx)];
  __syncthreads();
  for (int i = 0; i < 32; i += 8)
    out[(size_t)(bx + ty + i) * rows + (by + tx)] = (bf16_t)tile[tx][ty + i];
}

// ---------------------------------------------------------------------------
// Generic bf16 WMMA GEMM:  C[M x N] = A[M x K] @ B, with B given as
// Bt[N x K] (row-major transpose) so fragment gathers are contiguous.
//
// Block tile: BM=128 x BN=64, BK=32. 256 threads = 8 waves, wave grid 4x2,
// each wave computes 32x32 (2x2 tiles of v_wmma_f32_16x16x32_bf16).
// Double-buffered LDS; async DMA of tile i+1 overlaps WMMA on tile i.
// ---------------------------------------------------------------------------
#define BM 128
#define BN 64
#define BK 32

template <bool HAS_BIAS, bool HAS_GAMMA, bool PRELU, bool OUT_BF16>
__global__ __launch_bounds__(256)
void gemm_wmma_bf16(const bf16_t* __restrict__ A,    // [M x K]
                    const bf16_t* __restrict__ Bt,   // [N x K]
                    const float*  __restrict__ bias, // [N] or null
                    const float*  __restrict__ gamma,// [M x N] or null
                    const float*  __restrict__ prelu_a_ptr, // scalar or null
                    float*        __restrict__ Cf,   // f32 out (if !OUT_BF16)
                    bf16_t*       __restrict__ Cb,   // bf16 out (if OUT_BF16)
                    int M, int N, int K) {
  __shared__ bf16_t sA[2][BM * BK];  // 2 x 8 KB
  __shared__ bf16_t sB[2][BN * BK];  // 2 x 4 KB

  const int tid  = threadIdx.x;
  const int wave = tid >> 5;      // 0..7
  const int lane = tid & 31;
  const int wm   = wave >> 1;     // 0..3  (32-row slab)
  const int wn   = wave & 1;      // 0..1  (32-col slab)

  const size_t blockM = (size_t)blockIdx.y * BM;
  const size_t blockN = (size_t)blockIdx.x * BN;

  v8f acc[2][2] = {};

  const int lrow  = lane & 15;        // row (A) / col (B) within 16-tile
  const int khalf = (lane >> 4) * 8;  // which 8-element K chunk this lane holds

  // per-thread staging coordinates (fixed across the K loop)
  const int ar0 = (tid) >> 2,        ac0 = ((tid) & 3) * 8;        // A chunk 0
  const int ar1 = (tid + 256) >> 2,  ac1 = ((tid + 256) & 3) * 8;  // A chunk 1
  const int brr = tid >> 2,          bcc = (tid & 3) * 8;          // B chunk

  // stage one K-tile (column kk) into LDS buffer `buf`
  auto stage = [&](int buf, int kk) {
#if USE_ASYNC_LDS && defined(__gfx1250__)
    uint32_t l0 = (uint32_t)(uintptr_t)&sA[buf][ar0 * BK + ac0];
    const bf16_t* g0 = &A[(blockM + ar0) * (size_t)K + (kk + ac0)];
    asm volatile("global_load_async_to_lds_b128 %0, %1, off"
                 :: "v"(l0), "v"(g0) : "memory");
    uint32_t l1 = (uint32_t)(uintptr_t)&sA[buf][ar1 * BK + ac1];
    const bf16_t* g1 = &A[(blockM + ar1) * (size_t)K + (kk + ac1)];
    asm volatile("global_load_async_to_lds_b128 %0, %1, off"
                 :: "v"(l1), "v"(g1) : "memory");
    uint32_t l2 = (uint32_t)(uintptr_t)&sB[buf][brr * BK + bcc];
    const bf16_t* g2 = &Bt[(blockN + brr) * (size_t)K + (kk + bcc)];
    asm volatile("global_load_async_to_lds_b128 %0, %1, off"
                 :: "v"(l2), "v"(g2) : "memory");
#else
    *(v8bf*)&sA[buf][ar0 * BK + ac0] =
        *(const v8bf*)&A[(blockM + ar0) * (size_t)K + (kk + ac0)];
    *(v8bf*)&sA[buf][ar1 * BK + ac1] =
        *(const v8bf*)&A[(blockM + ar1) * (size_t)K + (kk + ac1)];
    *(v8bf*)&sB[buf][brr * BK + bcc] =
        *(const v8bf*)&Bt[(blockN + brr) * (size_t)K + (kk + bcc)];
#endif
  };

  // prologue: fill buffer 0 with the first K-tile
  stage(0, 0);
#if USE_ASYNC_LDS && defined(__gfx1250__)
  asm volatile("s_wait_asynccnt 0" ::: "memory");
#endif
  __syncthreads();

  for (int k0 = 0; k0 < K; k0 += BK) {
    const int buf = (k0 / BK) & 1;

    // kick off DMA of the next K-tile into the other buffer; it overlaps
    // the WMMA chain below. The barrier that ended the previous iteration
    // guarantees nobody is still reading buf^1.
    if (k0 + BK < K) {
      stage(buf ^ 1, k0 + BK);
      if (k0 + 2 * BK < K) {  // keep L2 warm two tiles ahead
        __builtin_prefetch(&A[(blockM + (tid >> 1)) * (size_t)K + (k0 + 2 * BK)], 0, 0);
        __builtin_prefetch(&Bt[(blockN + (tid & 63)) * (size_t)K + (k0 + 2 * BK)], 0, 0);
      }
    }

    // ---- fragments per ISA 16-bit layout:
    // lane l (l<16): row M=l, K = khalf..khalf+7 and khalf+16..khalf+23
    union { v16bf v; v8bf h[2]; } afrag[2], bfrag[2];
    #pragma unroll
    for (int i = 0; i < 2; ++i) {
      int ar = wm * 32 + i * 16 + lrow;
      afrag[i].h[0] = *(v8bf*)&sA[buf][ar * BK + khalf];
      afrag[i].h[1] = *(v8bf*)&sA[buf][ar * BK + khalf + 16];
      int br = wn * 32 + i * 16 + lrow;
      bfrag[i].h[0] = *(v8bf*)&sB[buf][br * BK + khalf];
      bfrag[i].h[1] = *(v8bf*)&sB[buf][br * BK + khalf + 16];
    }

    #pragma unroll
    for (int i = 0; i < 2; ++i)
      #pragma unroll
      for (int j = 0; j < 2; ++j)
        acc[i][j] = __builtin_amdgcn_wmma_f32_16x16x32_bf16(
            /*neg_a=*/false, afrag[i].v, /*neg_b=*/false, bfrag[j].v,
            /*c_mod=*/(short)0, acc[i][j], /*reuse_a=*/false, /*reuse_b=*/false);

    // drain this iteration's async DMA, then rendezvous: after this barrier
    // buf^1 is valid for everyone and buf may be overwritten next iteration.
#if USE_ASYNC_LDS && defined(__gfx1250__)
    asm volatile("s_wait_asynccnt 0" ::: "memory");
#endif
    __syncthreads();
  }

  // ---- epilogue. C/D layout: VGPR v, lane l -> M = v + 8*(l>=16), N = l%16
  const float pa   = PRELU ? *prelu_a_ptr : 0.0f;
  const int   rsel = (lane >> 4) * 8;
  #pragma unroll
  for (int i = 0; i < 2; ++i) {
    #pragma unroll
    for (int j = 0; j < 2; ++j) {
      size_t col = blockN + (size_t)wn * 32 + j * 16 + lrow;
      #pragma unroll
      for (int v = 0; v < 8; ++v) {
        size_t row = blockM + (size_t)wm * 32 + i * 16 + v + rsel;
        float val = acc[i][j][v];
        if (HAS_GAMMA) val *= gamma[row * (size_t)N + col];
        if (HAS_BIAS)  val += bias[col];
        if (PRELU)     val = (val >= 0.0f) ? val : pa * val;
        if (OUT_BF16)  Cb[row * (size_t)N + col] = (bf16_t)val;
        else           Cf[row * (size_t)N + col] = val;
      }
    }
  }
}

// ---------------------------------------------------------------------------
// Host-side launch sequence
// ---------------------------------------------------------------------------
extern "C" void kernel_launch(void* const* d_in, const int* in_sizes, int n_in,
                              void* d_out, int out_size, void* d_ws, size_t ws_size,
                              hipStream_t stream) {
  const float* x       = (const float*)d_in[0];
  const float* d_gamma = (const float*)d_in[1];
  const float* Wq      = (const float*)d_in[2];
  const float* bq      = (const float*)d_in[3];
  const float* Wk      = (const float*)d_in[4];
  const float* bk      = (const float*)d_in[5];
  const float* Wv      = (const float*)d_in[6];
  const float* bv      = (const float*)d_in[7];
  const float* Wr      = (const float*)d_in[8];
  const float* br      = (const float*)d_in[9];
  const float* prelu_a = (const float*)d_in[10];
  float* out = (float*)d_out;

  // workspace layout (bytes); total = 88 MB
  char* ws = (char*)d_ws;
  const size_t MB = 1024ull * 1024ull;
  bf16_t* xb  = (bf16_t*)(ws + 0);        //  8 MB : bf16 x          [T x IN]
  bf16_t* Wqt = (bf16_t*)(ws + 8  * MB);  //  2 MB : Wq^T bf16       [INTER x IN]
  bf16_t* Wkt = (bf16_t*)(ws + 10 * MB);  //  2 MB
  bf16_t* Wvt = (bf16_t*)(ws + 12 * MB);  //  2 MB
  bf16_t* Wrt = (bf16_t*)(ws + 14 * MB);  //  2 MB : Wr^T            [OUT x INTER]
  bf16_t* qb  = (bf16_t*)(ws + 16 * MB);  //  8 MB : q bf16          [T x INTER]
  bf16_t* kb  = (bf16_t*)(ws + 24 * MB);  //  8 MB : k bf16          [T x INTER]
  bf16_t* vb  = (bf16_t*)(ws + 32 * MB);  //  8 MB : v bf16          [T x INTER]
  bf16_t* vtb = (bf16_t*)(ws + 40 * MB);  //  8 MB : v^T bf16        [INTER x T]
  bf16_t* sb  = (bf16_t*)(ws + 48 * MB);  // 32 MB : gamma*scores    [T x T]
  bf16_t* rb  = (bf16_t*)(ws + 80 * MB);  //  8 MB : retained bf16   [T x INTER]

  dim3 blk1(256);
  dim3 tblk(32, 8);

  // 1) x -> bf16
  convert_f32_to_bf16<<<(T_DIM * IN_DIM / 4 + 255) / 256, blk1, 0, stream>>>(
      x, xb, T_DIM * IN_DIM / 4);

  // 2) weights -> bf16, transposed to [N x K]
  transpose_convert_bf16<float><<<dim3(INTER_DIM / 32, IN_DIM / 32), tblk, 0, stream>>>(
      Wq, Wqt, IN_DIM, INTER_DIM);
  transpose_convert_bf16<float><<<dim3(INTER_DIM / 32, IN_DIM / 32), tblk, 0, stream>>>(
      Wk, Wkt, IN_DIM, INTER_DIM);
  transpose_convert_bf16<float><<<dim3(INTER_DIM / 32, IN_DIM / 32), tblk, 0, stream>>>(
      Wv, Wvt, IN_DIM, INTER_DIM);
  transpose_convert_bf16<float><<<dim3(OUT_DIM / 32, INTER_DIM / 32), tblk, 0, stream>>>(
      Wr, Wrt, INTER_DIM, OUT_DIM);

  // 3) q/k/v projections (bias epilogue, bf16 out)
  dim3 gproj(INTER_DIM / BN, T_DIM / BM);
  gemm_wmma_bf16<true, false, false, true><<<gproj, blk1, 0, stream>>>(
      xb, Wqt, bq, nullptr, nullptr, nullptr, qb, T_DIM, INTER_DIM, IN_DIM);
  gemm_wmma_bf16<true, false, false, true><<<gproj, blk1, 0, stream>>>(
      xb, Wkt, bk, nullptr, nullptr, nullptr, kb, T_DIM, INTER_DIM, IN_DIM);
  gemm_wmma_bf16<true, false, false, true><<<gproj, blk1, 0, stream>>>(
      xb, Wvt, bv, nullptr, nullptr, nullptr, vb, T_DIM, INTER_DIM, IN_DIM);

  // 4) v^T for the retained GEMM's B operand
  transpose_convert_bf16<bf16_t><<<dim3(INTER_DIM / 32, T_DIM / 32), tblk, 0, stream>>>(
      vb, vtb, T_DIM, INTER_DIM);

  // 5) scores = q @ k^T with gamma mask epilogue (Bt == k, already [N x K])
  dim3 gscore(T_DIM / BN, T_DIM / BM);
  gemm_wmma_bf16<false, true, false, true><<<gscore, blk1, 0, stream>>>(
      qb, kb, nullptr, d_gamma, nullptr, nullptr, sb, T_DIM, T_DIM, INTER_DIM);

  // 6) retained = (gamma*scores) @ v
  dim3 gret(INTER_DIM / BN, T_DIM / BM);
  gemm_wmma_bf16<false, false, false, true><<<gret, blk1, 0, stream>>>(
      sb, vtb, nullptr, nullptr, nullptr, nullptr, rb, T_DIM, INTER_DIM, T_DIM);

  // 7) out = PReLU(retained @ Wr + br), f32 out
  dim3 gout(OUT_DIM / BN, T_DIM / BM);
  gemm_wmma_bf16<true, false, true, false><<<gout, blk1, 0, stream>>>(
      rb, Wrt, br, nullptr, prelu_a, out, nullptr, T_DIM, OUT_DIM, INTER_DIM);

  (void)in_sizes; (void)n_in; (void)out_size; (void)ws_size;
}
